// WuBuNestingModel_54546084659760
// MI455X (gfx1250) — matrix operand order repaired and checked by reading, not compiled, not run on hardware
//
#include <hip/hip_runtime.h>
#include <cmath>

typedef __attribute__((ext_vector_type(16))) __bf16 v16bf;
typedef __attribute__((ext_vector_type(8)))  __bf16 v8bf;
typedef __attribute__((ext_vector_type(8)))  float  v8f;

#define EPSR 1e-7f

// ---- small blob (biases + scales), float indices ----
enum : int {
  BC_O   = 0,     // 64  folded bias (b_in@W_tan + b_tan)
  T0B1_O = 64,    // 56
  T0B2_O = 120,   // 48
  T1B1_O = 168,   // 40
  T1B2_O = 208,   // 32
  BOUT_O = 240,   // 10
  SC_O   = 250,   // 3 (s0,s1,s2)
  BLOB_N = 256,
};

// ---- bf16 WMMA B-tile pack (element offsets), each row-major [N][64] ----
enum : int {
  WB_G0  = 0,      // 64x64 : n<56, k<64  R0^T@t0_w1
  WB_W2A = 4096,   // 64x64 : n<48, k<56  t0_w2
  WB_G1  = 8192,   // 64x64 : n<40, k<48  R1^T@t1_w1
  WB_W2B = 12288,  // 64x64 : n<32, k<40  t1_w2
  WB_O0  = 16384,  // 16x64 : n<10, k<64  W_out[0:64]
  WB_O1  = 17408,  // 16x64 : n<10, k<48  W_out[64:112]
  WB_O2  = 18432,  // 16x64 : n<10, k<32  W_out[112:144]
  WB_N   = 19456,
};

// ---- d_ws layout (float indices) ----
enum : int {
  WS_BLOB_F = 25600,            // after WcT bf16 (64x800 = 25600 floats)
  WS_R0_F   = WS_BLOB_F + BLOB_N,   // 25856 (64x64)
  WS_R1_F   = WS_R0_F + 4096,       // 29952 (48x48)
  WS_WB_F   = WS_R1_F + 2304,       // 32256 (19456 bf16 = 9728 floats)
};

// ---------------------------------------------------------------------------
// expm(A - A^T) via scaling-and-squaring + Taylor/Horner.
// ---------------------------------------------------------------------------
__global__ __launch_bounds__(256) void wubu_expm(const float* __restrict__ A0,
                                                 const float* __restrict__ A1,
                                                 float* __restrict__ R0,
                                                 float* __restrict__ R1) {
  __shared__ float S[64 * 64];
  __shared__ float P[64 * 64];
  __shared__ float T2[64 * 64];
  __shared__ float red[64];
  __shared__ float s_scl;
  __shared__ int   s_e;

  const int n = (blockIdx.x == 0) ? 64 : 48;
  const float* A = (blockIdx.x == 0) ? A0 : A1;
  float* R = (blockIdx.x == 0) ? R0 : R1;
  const int tid = threadIdx.x;
  const int nn = n * n;

  for (int idx = tid; idx < nn; idx += 256) {
    int r = idx / n, c = idx - r * n;
    S[idx] = A[r * n + c] - A[c * n + r];
  }
  __syncthreads();
  if (tid < n) {
    float s = 0.f;
    for (int c = 0; c < n; ++c) s += fabsf(S[tid * n + c]);
    red[tid] = s;
  }
  __syncthreads();
  if (tid == 0) {
    float m = 0.f;
    for (int i = 0; i < n; ++i) m = fmaxf(m, red[i]);
    int e = 0;
    while (m > 0.5f && e < 30) { m *= 0.5f; ++e; }
    s_e = e;
    s_scl = ldexpf(1.f, -e);
  }
  __syncthreads();
  const float scl = s_scl;
  const int   e   = s_e;
  for (int idx = tid; idx < nn; idx += 256) S[idx] *= scl;
  for (int idx = tid; idx < nn; idx += 256) {
    int r = idx / n, c = idx - r * n;
    P[idx] = (r == c) ? 1.f : 0.f;
  }
  __syncthreads();
  for (int k = 12; k >= 1; --k) {
    for (int idx = tid; idx < nn; idx += 256) {
      int r = idx / n, c = idx - r * n;
      float a = 0.f;
      for (int j = 0; j < n; ++j) a += S[r * n + j] * P[j * n + c];
      T2[idx] = a;
    }
    __syncthreads();
    const float inv = 1.f / (float)k;
    for (int idx = tid; idx < nn; idx += 256) {
      int r = idx / n, c = idx - r * n;
      P[idx] = ((r == c) ? 1.f : 0.f) + T2[idx] * inv;
    }
    __syncthreads();
  }
  for (int t = 0; t < e; ++t) {
    for (int idx = tid; idx < nn; idx += 256) {
      int r = idx / n, c = idx - r * n;
      float a = 0.f;
      for (int j = 0; j < n; ++j) a += P[r * n + j] * P[j * n + c];
      T2[idx] = a;
    }
    __syncthreads();
    for (int idx = tid; idx < nn; idx += 256) P[idx] = T2[idx];
    __syncthreads();
  }
  for (int idx = tid; idx < nn; idx += 256) R[idx] = P[idx];
}

// ---------------------------------------------------------------------------
// prep1: WcT = (W_in @ W_tan)^T in bf16 (K padded 784->800) + bias/scale blob.
// ---------------------------------------------------------------------------
__global__ __launch_bounds__(256) void wubu_prep1(
    const float* __restrict__ W_in,  const float* __restrict__ b_in,
    const float* __restrict__ W_tan, const float* __restrict__ b_tan,
    const float* __restrict__ t0_b1, const float* __restrict__ t0_b2,
    const float* __restrict__ t1_b1, const float* __restrict__ t1_b2,
    const float* __restrict__ b_out,
    const float* __restrict__ s0, const float* __restrict__ s1,
    const float* __restrict__ s2,
    __bf16* __restrict__ wct, float* __restrict__ blob) {
  const int tid = threadIdx.x;
  for (int idx = tid; idx < 64 * 800; idx += 256) {
    int j = idx / 800, i = idx - j * 800;
    float s = 0.f;
    if (i < 784) {
      for (int m = 0; m < 128; ++m) s += W_in[i * 128 + m] * W_tan[m * 64 + j];
    }
    wct[j * 800 + i] = (__bf16)s;
  }
  if (tid < 64) {
    float s = b_tan[tid];
    for (int m = 0; m < 128; ++m) s += b_in[m] * W_tan[m * 64 + tid];
    blob[BC_O + tid] = s;
  }
  if (tid < 56) blob[T0B1_O + tid] = t0_b1[tid];
  if (tid < 48) blob[T0B2_O + tid] = t0_b2[tid];
  if (tid < 40) blob[T1B1_O + tid] = t1_b1[tid];
  if (tid < 32) blob[T1B2_O + tid] = t1_b2[tid];
  if (tid < 10) blob[BOUT_O + tid] = b_out[tid];
  if (tid == 0) {
    blob[SC_O + 0] = s0[0];
    blob[SC_O + 1] = s1[0];
    blob[SC_O + 2] = s2[0];
  }
}

// ---------------------------------------------------------------------------
// prep2: build all bf16 WMMA B-tiles (rotations folded: G = R^T @ w1).
// Tiles are [N(pad)][64] row-major, zero padded in both N and K.
// ---------------------------------------------------------------------------
__global__ __launch_bounds__(256) void wubu_prep2(
    const float* __restrict__ t0_w1, const float* __restrict__ t0_w2,
    const float* __restrict__ t1_w1, const float* __restrict__ t1_w2,
    const float* __restrict__ W_out,
    const float* __restrict__ R0, const float* __restrict__ R1,
    __bf16* __restrict__ wbf) {
  const int tid = threadIdx.x;
  // G0 = R0^T @ t0_w1 : [k<64][n<56]
  for (int idx = tid; idx < 4096; idx += 256) {
    int n = idx >> 6, k = idx & 63;
    float s = 0.f;
    if (n < 56) {
      for (int m = 0; m < 64; ++m) s += R0[m * 64 + k] * t0_w1[m * 56 + n];
    }
    wbf[WB_G0 + idx] = (__bf16)s;
  }
  // t0_w2 : [k<56][n<48]
  for (int idx = tid; idx < 4096; idx += 256) {
    int n = idx >> 6, k = idx & 63;
    float s = (n < 48 && k < 56) ? t0_w2[k * 48 + n] : 0.f;
    wbf[WB_W2A + idx] = (__bf16)s;
  }
  // G1 = R1^T @ t1_w1 : [k<48][n<40]
  for (int idx = tid; idx < 4096; idx += 256) {
    int n = idx >> 6, k = idx & 63;
    float s = 0.f;
    if (n < 40 && k < 48) {
      for (int m = 0; m < 48; ++m) s += R1[m * 48 + k] * t1_w1[m * 40 + n];
    }
    wbf[WB_G1 + idx] = (__bf16)s;
  }
  // t1_w2 : [k<40][n<32]
  for (int idx = tid; idx < 4096; idx += 256) {
    int n = idx >> 6, k = idx & 63;
    float s = (n < 32 && k < 40) ? t1_w2[k * 32 + n] : 0.f;
    wbf[WB_W2B + idx] = (__bf16)s;
  }
  // W_out segments : [16][64]
  for (int idx = tid; idx < 1024; idx += 256) {
    int n = idx >> 6, k = idx & 63;
    wbf[WB_O0 + idx] = (__bf16)((n < 10) ? W_out[k * 10 + n] : 0.f);
    wbf[WB_O1 + idx] = (__bf16)((n < 10 && k < 48) ? W_out[(64 + k) * 10 + n] : 0.f);
    wbf[WB_O2 + idx] = (__bf16)((n < 10 && k < 32) ? W_out[(112 + k) * 10 + n] : 0.f);
  }
}

// ---------------------------------------------------------------------------
// expmap0 -> projx -> logmap0 collapses to one scalar per row vector.
// ---------------------------------------------------------------------------
__device__ __forceinline__ float level_coef(const float* v, int d, float c,
                                            float scale, float smin) {
  float s = fmaxf(scale, smin);
  float nn = 0.f;
  for (int k = 0; k < d; ++k) { float u = v[k] * s; nn += u * u; }
  float un  = fmaxf(sqrtf(nn), EPSR);
  float sqc = sqrtf(c);
  float th  = tanhf(sqc * un);
  float coefX = th / (sqc * un) * s;
  float xn  = th / sqc;
  float maxn = (1.f - 1e-5f) / sqc;
  float fac = fminf(1.f, maxn / fmaxf(xn, EPSR));
  coefX *= fac;
  float xn2 = fmaxf(xn * fac, EPSR);
  float arg = fminf(fmaxf(sqc * xn2, 0.f), 1.f - 1e-7f);
  float coefT = atanhf(arg) / (sqc * xn2);
  return coefT * coefX / (s + EPSR);
}

// ---- WMMA fragment loaders (stride-64 bf16 row-major tiles in LDS) ----
__device__ __forceinline__ v16bf ldA64(const __bf16* aT, int row, int kstep,
                                       bool hi) {
  const __bf16* p = aT + row * 64 + kstep * 32 + (hi ? 8 : 0);
  v8bf lo = *(const v8bf*)p;
  v8bf hv = *(const v8bf*)(p + 16);
  return __builtin_shufflevector(lo, hv, 0, 1, 2, 3, 4, 5, 6, 7,
                                 8, 9, 10, 11, 12, 13, 14, 15);
}
__device__ __forceinline__ v16bf ldB64(const __bf16* wb, int n, int l15,
                                       int kstep, bool hi) {
  const __bf16* p = wb + (n * 16 + l15) * 64 + kstep * 32 + (hi ? 8 : 0);
  v8bf lo = *(const v8bf*)p;
  v8bf hv = *(const v8bf*)(p + 16);
  return __builtin_shufflevector(lo, hv, 0, 1, 2, 3, 4, 5, 6, 7,
                                 8, 9, 10, 11, 12, 13, 14, 15);
}
#define WMMA_BF16(A, B, C) __builtin_amdgcn_wmma_f32_16x16x32_bf16( \
    false, (A), false, (B), (short)0, (C), false, false)

// ---------------------------------------------------------------------------
// Main fused kernel.
// ---------------------------------------------------------------------------
__global__ __launch_bounds__(256) void wubu_main(const float* __restrict__ x,
                                                 const __bf16* __restrict__ wct,
                                                 const float* __restrict__ blob,
                                                 const __bf16* __restrict__ wbf,
                                                 float* __restrict__ out,
                                                 int Brows) {
  __shared__ __align__(16) __bf16 xTile[128 * 32];   // 8 KB
  __shared__ __align__(16) __bf16 wTile[64 * 32];    // 4 KB
  __shared__ __align__(16) __bf16 aT[128 * 64];      // 16 KB staged A tiles
  __shared__ __align__(16) __bf16 wbAll[WB_N];       // 38 KB bf16 B tiles
  __shared__ float smw[BLOB_N];                      // 1 KB biases/scales
  __shared__ float vT[128 * 65];                     // 33.3 KB f32 vectors
  __shared__ float coefR[128];

  const int tid  = threadIdx.x;
  const int lane = tid & 31;
  const int wave = tid >> 5;
  const int l15  = lane & 15;
  const bool hiHalf = lane >= 16;
  const size_t blockRow = (size_t)blockIdx.x * 128;

  // stage constants into LDS (covered by first barrier in GEMM loop)
  if (tid < BLOB_N) smw[tid] = blob[tid];
  {
    const v8bf* src = (const v8bf*)wbf;
    v8bf* dst = (v8bf*)wbAll;
    for (int i = tid; i < WB_N / 8; i += 256) dst[i] = src[i];
  }

  v8f acc[4];
#pragma unroll
  for (int n = 0; n < 4; ++n)
    acc[n] = (v8f){0.f, 0.f, 0.f, 0.f, 0.f, 0.f, 0.f, 0.f};

  // x-tile loader: 2 threads/row, 16 cols each; 784 = 49*16 so the per-16-col
  // segment validity is a single predicate.
  const int xr  = tid >> 1;
  const int xcH = (tid & 1) * 16;
  const size_t xrow = blockRow + xr;
  const bool rowOk = xrow < (size_t)Brows;
  const float* xbase = x + xrow * 784;

  // WcT async-loader: 1 thread per 16B, DMA straight into LDS.
  const int wr = tid >> 2;
  const int wc = (tid & 3) * 8;
  const unsigned wLds = (unsigned)(uintptr_t)(&wTile[wr * 32 + wc]);
  const __bf16* wGlob = wct + wr * 800 + wc;

  // ---- main GEMM: v = x @ Wc, K=784 (pad 800), BK=32 ----
  for (int kt = 0; kt < 25; ++kt) {
    const int k0 = kt * 32;
    {
      const unsigned long long ga = (unsigned long long)(uintptr_t)(wGlob + k0);
      asm volatile("global_load_async_to_lds_b128 %0, %1, off"
                   :: "v"(wLds), "v"(ga) : "memory");
    }
    {
      const bool ok = rowOk && (k0 + xcH < 784);
      float4 f0 = make_float4(0.f, 0.f, 0.f, 0.f), f1 = f0, f2 = f0, f3 = f0;
      if (ok) {
        const float4* xp4 = (const float4*)(xbase + k0 + xcH);
        f0 = xp4[0]; f1 = xp4[1]; f2 = xp4[2]; f3 = xp4[3];
        if (kt + 1 < 25)
          __builtin_prefetch((const float*)xp4 + 32, 0, 3);
      }
      v8bf b0 = { (__bf16)f0.x, (__bf16)f0.y, (__bf16)f0.z, (__bf16)f0.w,
                  (__bf16)f1.x, (__bf16)f1.y, (__bf16)f1.z, (__bf16)f1.w };
      v8bf b1 = { (__bf16)f2.x, (__bf16)f2.y, (__bf16)f2.z, (__bf16)f2.w,
                  (__bf16)f3.x, (__bf16)f3.y, (__bf16)f3.z, (__bf16)f3.w };
      *(v8bf*)(&xTile[xr * 32 + xcH])     = b0;
      *(v8bf*)(&xTile[xr * 32 + xcH + 8]) = b1;
    }
    asm volatile("s_wait_asynccnt 0x0" ::: "memory");
    __syncthreads();
    {
      const int aRow = wave * 16 + l15;
      const int eoff = hiHalf ? 8 : 0;
      const __bf16* ap = &xTile[aRow * 32 + eoff];
      v8bf alo = *(const v8bf*)ap;
      v8bf ahi = *(const v8bf*)(ap + 16);
      v16bf aF = __builtin_shufflevector(alo, ahi, 0, 1, 2, 3, 4, 5, 6, 7,
                                         8, 9, 10, 11, 12, 13, 14, 15);
#pragma unroll
      for (int n = 0; n < 4; ++n) {
        const __bf16* bp = &wTile[(n * 16 + l15) * 32 + (hiHalf ? 8 : 0)];
        v8bf blo = *(const v8bf*)bp;
        v8bf bhi = *(const v8bf*)(bp + 16);
        v16bf bF = __builtin_shufflevector(blo, bhi, 0, 1, 2, 3, 4, 5, 6, 7,
                                           8, 9, 10, 11, 12, 13, 14, 15);
        acc[n] = WMMA_BF16(aF, bF, acc[n]);
      }
    }
    __syncthreads();
  }

  // accumulators -> vT (C layout: VGPR r -> M=r / r+8 by lane half)
#pragma unroll
  for (int n = 0; n < 4; ++n) {
#pragma unroll
    for (int r = 0; r < 8; ++r) {
      const int rowL = wave * 16 + r + (hiHalf ? 8 : 0);
      vT[rowL * 65 + n * 16 + l15] = acc[n][r];
    }
  }
  __syncthreads();

  const float sc0 = smw[SC_O + 0], sc1 = smw[SC_O + 1], sc2 = smw[SC_O + 2];
  const int strip = wave * 16;
  const int rofs  = hiHalf ? 8 : 0;

  // ---- level 0 coefficient + A staging (coef baked into bf16 A tile) ----
  if (tid < 128) {
    float* vrow = &vT[tid * 65];
    float cf = 0.f;
    if (blockRow + tid < (size_t)Brows) {
      for (int k = 0; k < 64; ++k) vrow[k] += smw[BC_O + k];
      cf = level_coef(vrow, 64, 1.0f, sc0, 1e-4f);
    }
    coefR[tid] = cf;
  }
  __syncthreads();
  for (int idx = tid; idx < 128 * 64; idx += 256) {
    int row = idx >> 6, k = idx & 63;
    aT[idx] = (__bf16)(coefR[row] * vT[row * 65 + k]);
  }
  __syncthreads();

  v8f oAcc = (v8f){0.f, 0.f, 0.f, 0.f, 0.f, 0.f, 0.f, 0.f};
  const v8f Z8 = oAcc;

  // tout0 @ Wout[0:64] and H = relu(tout0 @ G0 + b1)
  {
    v16bf aF0 = ldA64(aT, strip + l15, 0, hiHalf);
    v16bf aF1 = ldA64(aT, strip + l15, 1, hiHalf);
    oAcc = WMMA_BF16(aF0, ldB64(&wbAll[WB_O0], 0, l15, 0, hiHalf), oAcc);
    oAcc = WMMA_BF16(aF1, ldB64(&wbAll[WB_O0], 0, l15, 1, hiHalf), oAcc);
    v8f hA[4];
#pragma unroll
    for (int n = 0; n < 4; ++n) {
      hA[n] = WMMA_BF16(aF0, ldB64(&wbAll[WB_G0], n, l15, 0, hiHalf), Z8);
      hA[n] = WMMA_BF16(aF1, ldB64(&wbAll[WB_G0], n, l15, 1, hiHalf), hA[n]);
    }
#pragma unroll
    for (int n = 0; n < 4; ++n) {
#pragma unroll
      for (int r = 0; r < 8; ++r) {
        const int row = strip + r + rofs;
        const int col = n * 16 + l15;
        float v = (col < 56) ? fmaxf(hA[n][r] + smw[T0B1_O + col], 0.f) : 0.f;
        aT[row * 64 + col] = (__bf16)v;
      }
    }
  }
  // V1 = H @ t0_w2 + b2 (f32 into vT, cols<48)
  {
    v16bf aF0 = ldA64(aT, strip + l15, 0, hiHalf);
    v16bf aF1 = ldA64(aT, strip + l15, 1, hiHalf);
#pragma unroll
    for (int n = 0; n < 3; ++n) {
      v8f vA = WMMA_BF16(aF0, ldB64(&wbAll[WB_W2A], n, l15, 0, hiHalf), Z8);
      vA = WMMA_BF16(aF1, ldB64(&wbAll[WB_W2A], n, l15, 1, hiHalf), vA);
#pragma unroll
      for (int r = 0; r < 8; ++r) {
        const int row = strip + r + rofs;
        const int col = n * 16 + l15;
        vT[row * 65 + col] = vA[r] + smw[T0B2_O + col];
      }
    }
  }
  __syncthreads();

  // ---- level 1 ----
  if (tid < 128) {
    coefR[tid] = (blockRow + tid < (size_t)Brows)
                     ? level_coef(&vT[tid * 65], 48, 0.8f, sc1, 1e-5f) : 0.f;
  }
  __syncthreads();
  for (int idx = tid; idx < 128 * 64; idx += 256) {
    int row = idx >> 6, k = idx & 63;
    float v = (k < 48) ? coefR[row] * vT[row * 65 + k] : 0.f;
    aT[idx] = (__bf16)v;
  }
  __syncthreads();
  // tout1 @ Wout[64:112] and H2 = relu(tout1 @ G1 + b1)
  {
    v16bf aF0 = ldA64(aT, strip + l15, 0, hiHalf);
    v16bf aF1 = ldA64(aT, strip + l15, 1, hiHalf);
    oAcc = WMMA_BF16(aF0, ldB64(&wbAll[WB_O1], 0, l15, 0, hiHalf), oAcc);
    oAcc = WMMA_BF16(aF1, ldB64(&wbAll[WB_O1], 0, l15, 1, hiHalf), oAcc);
    v8f hA[3];
#pragma unroll
    for (int n = 0; n < 3; ++n) {
      hA[n] = WMMA_BF16(aF0, ldB64(&wbAll[WB_G1], n, l15, 0, hiHalf), Z8);
      hA[n] = WMMA_BF16(aF1, ldB64(&wbAll[WB_G1], n, l15, 1, hiHalf), hA[n]);
    }
#pragma unroll
    for (int n = 0; n < 3; ++n) {
#pragma unroll
      for (int r = 0; r < 8; ++r) {
        const int row = strip + r + rofs;
        const int col = n * 16 + l15;
        float v = (col < 40) ? fmaxf(hA[n][r] + smw[T1B1_O + col], 0.f) : 0.f;
        aT[row * 64 + col] = (__bf16)v;
      }
    }
    // cols 48..63 of aT keep level-1 staging zeros -> K padding stays valid
  }
  // V2 = H2 @ t1_w2 + b2 (f32 into vT, cols<32)
  {
    v16bf aF0 = ldA64(aT, strip + l15, 0, hiHalf);
    v16bf aF1 = ldA64(aT, strip + l15, 1, hiHalf);
#pragma unroll
    for (int n = 0; n < 2; ++n) {
      v8f vA = WMMA_BF16(aF0, ldB64(&wbAll[WB_W2B], n, l15, 0, hiHalf), Z8);
      vA = WMMA_BF16(aF1, ldB64(&wbAll[WB_W2B], n, l15, 1, hiHalf), vA);
#pragma unroll
      for (int r = 0; r < 8; ++r) {
        const int row = strip + r + rofs;
        const int col = n * 16 + l15;
        vT[row * 65 + col] = vA[r] + smw[T1B2_O + col];
      }
    }
  }
  __syncthreads();

  // ---- level 2 ----
  if (tid < 128) {
    coefR[tid] = (blockRow + tid < (size_t)Brows)
                     ? level_coef(&vT[tid * 65], 32, 1.2f, sc2, 1e-6f) : 0.f;
  }
  __syncthreads();
  for (int idx = tid; idx < 128 * 64; idx += 256) {
    int row = idx >> 6, k = idx & 63;
    float v = (k < 32) ? coefR[row] * vT[row * 65 + k] : 0.f;
    aT[idx] = (__bf16)v;
  }
  __syncthreads();
  {
    v16bf aF0 = ldA64(aT, strip + l15, 0, hiHalf);
    oAcc = WMMA_BF16(aF0, ldB64(&wbAll[WB_O2], 0, l15, 0, hiHalf), oAcc);
  }

  // ---- final store: oAcc row-major, N=lane (cols 10..15 are zero pad) ----
  if (l15 < 10) {
    const float bo = smw[BOUT_O + l15];
#pragma unroll
    for (int r = 0; r < 8; ++r) {
      const size_t row = blockRow + strip + r + rofs;
      if (row < (size_t)Brows) out[row * 10 + l15] = oAcc[r] + bo;
    }
  }
}

// ---------------------------------------------------------------------------
extern "C" void kernel_launch(void* const* d_in, const int* in_sizes, int n_in,
                              void* d_out, int out_size, void* d_ws, size_t ws_size,
                              hipStream_t stream) {
  (void)n_in; (void)out_size; (void)ws_size;
  const float* x      = (const float*)d_in[0];
  const float* W_in   = (const float*)d_in[1];
  const float* b_in   = (const float*)d_in[2];
  const float* W_tan  = (const float*)d_in[3];
  const float* b_tan  = (const float*)d_in[4];
  const float* s0     = (const float*)d_in[5];
  const float* s1     = (const float*)d_in[6];
  const float* s2     = (const float*)d_in[7];
  // d_in[8]=tp0, d_in[9]=tp1: boundary side-path, does not affect output
  const float* A0     = (const float*)d_in[10];
  const float* A1     = (const float*)d_in[11];
  const float* t0_w1  = (const float*)d_in[12];
  const float* t0_b1  = (const float*)d_in[13];
  const float* t0_w2  = (const float*)d_in[14];
  const float* t0_b2  = (const float*)d_in[15];
  const float* t1_w1  = (const float*)d_in[16];
  const float* t1_b1  = (const float*)d_in[17];
  const float* t1_w2  = (const float*)d_in[18];
  const float* t1_b2  = (const float*)d_in[19];
  const float* W_out  = (const float*)d_in[20];
  const float* b_out  = (const float*)d_in[21];

  const int Brows = in_sizes[0] / 784;
  float*  wsF  = (float*)d_ws;
  __bf16* wct  = (__bf16*)d_ws;
  float*  blob = wsF + WS_BLOB_F;
  float*  R0   = wsF + WS_R0_F;
  float*  R1   = wsF + WS_R1_F;
  __bf16* wbf  = (__bf16*)(wsF + WS_WB_F);

  wubu_expm<<<2, 256, 0, stream>>>(A0, A1, R0, R1);
  wubu_prep1<<<1, 256, 0, stream>>>(W_in, b_in, W_tan, b_tan,
                                    t0_b1, t0_b2, t1_b1, t1_b2, b_out,
                                    s0, s1, s2, wct, blob);
  wubu_prep2<<<1, 256, 0, stream>>>(t0_w1, t0_w2, t1_w1, t1_w2, W_out,
                                    R0, R1, wbf);

  const int grid = (Brows + 127) / 128;
  wubu_main<<<grid, 256, 0, stream>>>(x, wct, blob, wbf, (float*)d_out, Brows);
}
